// OCAB_71863392797254
// MI455X (gfx1250) — matrix operand order repaired and checked by reading, not compiled, not run on hardware
//
#include <hip/hip_runtime.h>
#include <math.h>

// ---------------------------------------------------------------------------
// Types / helpers
// ---------------------------------------------------------------------------
typedef _Float16 h8   __attribute__((ext_vector_type(8)));
typedef _Float16 v16h __attribute__((ext_vector_type(16)));
typedef float    v8f  __attribute__((ext_vector_type(8)));

#define SHUF16(lo, hi) __builtin_shufflevector(lo, hi, 0,1,2,3,4,5,6,7,8,9,10,11,12,13,14,15)

__device__ __forceinline__ v8f wmma_f16f32(v16h a, v16h b, v8f c) {
  return __builtin_amdgcn_wmma_f32_16x16x32_f16(false, a, false, b, (short)0, c, false, false);
}

__device__ __forceinline__ v8f v8f_zero() {
  v8f z;
#pragma unroll
  for (int i = 0; i < 8; ++i) z[i] = 0.0f;
  return z;
}

// Async global -> LDS copy, 16 bytes per lane. The VDST operand holds the
// LDS byte address = low 32 bits of the flat shared-memory address
// Tracked by ASYNCcnt.
__device__ __forceinline__ void async_g2l_b128(const void* gptr, void* lptr) {
  unsigned l = (unsigned)(size_t)lptr;
  asm volatile("global_load_async_to_lds_b128 %0, %1, off"
               :: "v"(l), "v"(gptr) : "memory");
}
__device__ __forceinline__ void wait_async0() {
  asm volatile("s_wait_asynccnt 0x0" ::: "memory");
}
__device__ __forceinline__ void wait_async3() {
  asm volatile("s_wait_asynccnt 0x3" ::: "memory");
}

#define LTOT     16384        // H*W
#define CDIM     192
#define INV_CNT  (1.0f / 3145728.0f)   // 1/(L*C)
#define EPSV     1e-6f
#define QKSCALE  0.17677669529663687f  // 32^-0.5

// ---------------------------------------------------------------------------
// Stats: per-batch sum / sumsq over (L, C)
// ---------------------------------------------------------------------------
__global__ void zero_stats_k(float* s) {
  if (threadIdx.x < 8) s[threadIdx.x] = 0.0f;
}

__global__ __launch_bounds__(256) void iln_stats_k(const float* __restrict__ x,
                                                   float* __restrict__ stats, int statOff) {
  int b = blockIdx.y;
  const float* p = x + (size_t)b * (LTOT * CDIM);
  float s = 0.0f, q = 0.0f;
  for (int i = blockIdx.x * blockDim.x + threadIdx.x; i < LTOT * CDIM;
       i += gridDim.x * blockDim.x) {
    float v = p[i];
    s += v;
    q += v * v;
  }
  __shared__ float ls[256], lq[256];
  int tid = threadIdx.x;
  ls[tid] = s; lq[tid] = q;
  __syncthreads();
  for (int off = 128; off > 0; off >>= 1) {
    if (tid < off) { ls[tid] += ls[tid + off]; lq[tid] += lq[tid + off]; }
    __syncthreads();
  }
  if (tid == 0) {
    atomicAdd(&stats[statOff + b * 2 + 0], ls[0]);
    atomicAdd(&stats[statOff + b * 2 + 1], lq[0]);
  }
}

// LN apply -> f16
__global__ __launch_bounds__(256) void iln_apply_k(const float* __restrict__ x,
                                                   const float* __restrict__ stats, int statOff,
                                                   const float* __restrict__ w,
                                                   const float* __restrict__ bb,
                                                   _Float16* __restrict__ out, int total) {
  int i = blockIdx.x * 256 + threadIdx.x;
  if (i >= total) return;
  int b = i / (LTOT * CDIM);
  int c = i % CDIM;
  float sum = stats[statOff + b * 2 + 0];
  float sq  = stats[statOff + b * 2 + 1];
  float mean = sum * INV_CNT;
  float var  = sq * INV_CNT - mean * mean;
  float sc   = fminf(sqrtf(var + EPSV), 2.0f);
  out[i] = (_Float16)(((x[i] - mean) / sc) * w[c] + bb[c]);
}

// one-time weight conversion f32 -> f16
__global__ __launch_bounds__(256) void f32_to_f16_k(const float* __restrict__ src,
                                                    _Float16* __restrict__ dst, int n) {
  int i = blockIdx.x * 256 + threadIdx.x;
  if (i < n) dst[i] = (_Float16)src[i];
}

// ---------------------------------------------------------------------------
// Generic WMMA GEMM:  D[M,N] = A[M,K](f16) * Bh[N,K]^T(f16) + bias
// mode 0: store f16 to outH
// mode 1: outF = resid + min(std,2)*ls[col]*val   (std from stats)
// Block tile: 128(M) x 64(N), 256 threads = 8 waves, K-step 32.
// Double-buffered async global->LDS staging (3 async instrs per wave/step).
// ---------------------------------------------------------------------------
__global__ __launch_bounds__(256) void gemm_wmma_k(const _Float16* __restrict__ A,
                                                   const _Float16* __restrict__ Bh,
                                                   const float* __restrict__ bias,
                                                   int N, int K,
                                                   _Float16* __restrict__ outH,
                                                   float* __restrict__ outF,
                                                   const float* __restrict__ resid,
                                                   const float* __restrict__ lsv,
                                                   const float* __restrict__ stats,
                                                   int statOff, int mode) {
  __shared__ _Float16 sA[2][128 * 32];
  __shared__ _Float16 sB[2][64 * 32];

  int tid  = threadIdx.x;
  int lane = tid & 31;
  int wave = tid >> 5;
  int m16  = lane & 15;
  int kh   = lane >> 4;
  int bn   = blockIdx.x * 64;
  int bm   = blockIdx.y * 128;

  // per-thread staging assignment
  int rowA = tid >> 1, segA = tid & 1;   // A: 128 rows x 32 halves, 32B/thread
  int nB   = tid >> 2, gB   = tid & 3;   // B: 64 rows x 32 halves, 16B/thread
  const _Float16* aBase = A  + (size_t)(bm + rowA) * K + segA * 16;
  const _Float16* bBase = Bh + (size_t)(bn + nB) * K + gB * 8;
  _Float16* aDst0 = &sA[0][rowA * 32 + segA * 16];
  _Float16* aDst1 = &sA[1][rowA * 32 + segA * 16];
  _Float16* bDst0 = &sB[0][nB * 32 + gB * 8];
  _Float16* bDst1 = &sB[1][nB * 32 + gB * 8];

  v8f acc[4];
#pragma unroll
  for (int i = 0; i < 4; ++i) acc[i] = v8f_zero();

  // prologue: stage step 0 into buffer 0
  async_g2l_b128(aBase,     aDst0);
  async_g2l_b128(aBase + 8, aDst0 + 8);
  async_g2l_b128(bBase,     bDst0);

  int nsteps = K >> 5;
  for (int s = 0; s < nsteps; ++s) {
    int p = s & 1;
    __syncthreads();  // all waves done computing from buf[1-p]; safe to refill
    if (s + 1 < nsteps) {
      int k0 = (s + 1) << 5;
      async_g2l_b128(aBase + k0,     p ? aDst0 : aDst1);
      async_g2l_b128(aBase + k0 + 8, (p ? aDst0 : aDst1) + 8);
      async_g2l_b128(bBase + k0,     p ? bDst0 : bDst1);
      wait_async3();  // step-s copies (3 older instrs) have landed
    } else {
      wait_async0();
    }
    __syncthreads();  // every wave's step-s data visible

    const _Float16* cA = sA[p];
    const _Float16* cB = sB[p];

    int arow = wave * 16 + m16;
    h8 alo = *(const h8*)(cA + arow * 32 + kh * 16);
    h8 ahi = *(const h8*)(cA + arow * 32 + kh * 16 + 8);
    v16h af = SHUF16(alo, ahi);

#pragma unroll
    for (int nt = 0; nt < 4; ++nt) {
      int brow = nt * 16 + m16;
      h8 blo = *(const h8*)(cB + brow * 32 + kh * 16);
      h8 bhi = *(const h8*)(cB + brow * 32 + kh * 16 + 8);
      v16h bf = SHUF16(blo, bhi);
      acc[nt] = wmma_f16f32(af, bf, acc[nt]);
    }
  }

  // --- epilogue ---
#pragma unroll
  for (int nt = 0; nt < 4; ++nt) {
#pragma unroll
    for (int r = 0; r < 8; ++r) {
      int row = bm + wave * 16 + r + 8 * kh;
      int col = bn + nt * 16 + m16;
      float v = acc[nt][r] + bias[col];
      if (mode == 0) {
        outH[(size_t)row * N + col] = (_Float16)v;
      } else {
        int b = row >> 14;  // 16384 rows per batch
        float sum = stats[statOff + b * 2 + 0];
        float sq  = stats[statOff + b * 2 + 1];
        float mean = sum * INV_CNT;
        float var  = sq * INV_CNT - mean * mean;
        float sc   = fminf(sqrtf(var + EPSV), 2.0f);
        outF[(size_t)row * N + col] =
            resid[(size_t)row * N + col] + sc * lsv[col] * v;
      }
    }
  }
}

// ---------------------------------------------------------------------------
// Overlapping-window attention, flash-style online softmax.
// One block per (window, head). 256 threads = 8 waves; each wave owns two
// 16-row query tiles. K/V staged in LDS 192 keys at a time (3 chunks x 12
// key-tiles). QK^T: one WMMA K-step (HD=32). PV: K in 32-key pairs.
// Q and K rows staged with async global->LDS; V transposed via ds scatter.
// ---------------------------------------------------------------------------
__global__ __launch_bounds__(256) void oca_attn_k(const _Float16* __restrict__ qkv,
                                                  const float* __restrict__ rpb,
                                                  _Float16* __restrict__ attn) {
  __shared__ _Float16 sQ[256 * 32];   // 16 KB  (row = q index)
  __shared__ _Float16 sK[192 * 32];   // 12 KB  (row = key, chunk-local)
  __shared__ _Float16 sVT[32 * 192];  // 12 KB  (row = d, col = key)
  __shared__ _Float16 sP[8 * 512];    //  8 KB  (per-wave 16x32 P slab)

  int tid  = threadIdx.x;
  int lane = tid & 31;
  int wave = tid >> 5;
  int m16  = lane & 15;
  int kh   = lane >> 4;

  int head = blockIdx.x % 6;
  int win  = blockIdx.x / 6;
  int b  = win >> 6;
  int wy = (win >> 3) & 7;
  int wx = win & 7;

  // --- stage Q window (q index == tid) via async copy ---
  {
    int qy = tid >> 4, qx = tid & 15;
    size_t rowG = (size_t)b * LTOT + (size_t)(wy * 16 + qy) * 128 + (wx * 16 + qx);
    const _Float16* src = qkv + rowG * 576 + head * 32;
    _Float16* d = sQ + tid * 32;
    async_g2l_b128(src,      d);
    async_g2l_b128(src + 8,  d + 8);
    async_g2l_b128(src + 16, d + 16);
    async_g2l_b128(src + 24, d + 24);
  }

  // per-wave flash state (2 M-tiles)
  float mrow[2][8], lrow[2][8];
  v8f o0[2], o1[2];
#pragma unroll
  for (int mi = 0; mi < 2; ++mi) {
#pragma unroll
    for (int r = 0; r < 8; ++r) { mrow[mi][r] = -3e38f; lrow[mi][r] = 0.0f; }
    o0[mi] = v8f_zero();
    o1[mi] = v8f_zero();
  }

  for (int c = 0; c < 3; ++c) {
    __syncthreads();  // previous chunk fully consumed

    // --- stage K/V chunk: keys [c*192, c*192+192) ---
    if (tid < 192) {
      int kg = c * 192 + tid;
      int ky = kg / 24, kx = kg % 24;
      int gy = wy * 16 - 4 + ky;
      int gx = wx * 16 - 4 + kx;
      bool ok = (gy >= 0) && (gy < 128) && (gx >= 0) && (gx < 128);
      if (ok) {
        size_t rowG = (size_t)b * LTOT + (size_t)gy * 128 + gx;
        const _Float16* ksrc = qkv + rowG * 576 + 192 + head * 32;
        const _Float16* vsrc = qkv + rowG * 576 + 384 + head * 32;
        _Float16* d = sK + tid * 32;
        async_g2l_b128(ksrc,      d);
        async_g2l_b128(ksrc + 8,  d + 8);
        async_g2l_b128(ksrc + 16, d + 16);
        async_g2l_b128(ksrc + 24, d + 24);
#pragma unroll
        for (int dd = 0; dd < 32; ++dd) sVT[dd * 192 + tid] = vsrc[dd];
      } else {
        _Float16 zz = (_Float16)0.0f;
        h8 z8 = {zz, zz, zz, zz, zz, zz, zz, zz};
        _Float16* d = sK + tid * 32;
        *(h8*)(d) = z8; *(h8*)(d + 8) = z8; *(h8*)(d + 16) = z8; *(h8*)(d + 24) = z8;
#pragma unroll
        for (int dd = 0; dd < 32; ++dd) sVT[dd * 192 + tid] = zz;
      }
    }
    wait_async0();   // covers Q asyncs (chunk 0) + this chunk's K asyncs
    __syncthreads();

    for (int mi = 0; mi < 2; ++mi) {
      int mt = wave * 2 + mi;
      // Q A-fragment for this M tile
      int arow = mt * 16 + m16;
      h8 qlo = *(const h8*)(sQ + arow * 32 + kh * 16);
      h8 qhi = *(const h8*)(sQ + arow * 32 + kh * 16 + 8);
      v16h aQ = SHUF16(qlo, qhi);

      for (int pr = 0; pr < 6; ++pr) {
        int ntl = pr * 2;  // chunk-local key tile (pair)

        // ---- S tiles: q @ k^T ----
        v8f s0, s1;
        {
          int key = ntl * 16 + m16;
          h8 blo = *(const h8*)(sK + key * 32 + kh * 16);
          h8 bhi = *(const h8*)(sK + key * 32 + kh * 16 + 8);
          v16h bK = SHUF16(blo, bhi);
          s0 = wmma_f16f32(aQ, bK, v8f_zero());
        }
        {
          int key = (ntl + 1) * 16 + m16;
          h8 blo = *(const h8*)(sK + key * 32 + kh * 16);
          h8 bhi = *(const h8*)(sK + key * 32 + kh * 16 + 8);
          v16h bK = SHUF16(blo, bhi);
          s1 = wmma_f16f32(aQ, bK, v8f_zero());
        }

        // ---- scale + relative-position bias ----
#pragma unroll
        for (int r = 0; r < 8; ++r) {
          int m  = mt * 16 + r + 8 * kh;
          int qy = m >> 4, qx = m & 15;
          int n0 = c * 192 + ntl * 16 + m16;
          int ky0 = n0 / 24, kx0 = n0 % 24;
          int idx0 = (ky0 - qy + 15) * 39 + (kx0 - qx + 15);
          s0[r] = s0[r] * QKSCALE + rpb[idx0 * 6 + head];
          int n1 = n0 + 16;
          int ky1 = n1 / 24, kx1 = n1 % 24;
          int idx1 = (ky1 - qy + 15) * 39 + (kx1 - qx + 15);
          s1[r] = s1[r] * QKSCALE + rpb[idx1 * 6 + head];
        }

        // ---- row max over 32 keys (16 lanes x 2 cols) ----
        float tmax[8];
#pragma unroll
        for (int r = 0; r < 8; ++r) {
          float t = fmaxf(s0[r], s1[r]);
          t = fmaxf(t, __shfl_xor(t, 1));
          t = fmaxf(t, __shfl_xor(t, 2));
          t = fmaxf(t, __shfl_xor(t, 4));
          t = fmaxf(t, __shfl_xor(t, 8));
          tmax[r] = t;
        }
        float fs[8];
#pragma unroll
        for (int r = 0; r < 8; ++r) {
          float nm = fmaxf(mrow[mi][r], tmax[r]);
          fs[r] = __expf(mrow[mi][r] - nm);
          mrow[mi][r] = nm;
        }
#pragma unroll
        for (int r = 0; r < 8; ++r) { o0[mi][r] *= fs[r]; o1[mi][r] *= fs[r]; }

        // ---- P = exp(S - m), row sums, store P to per-wave LDS slab ----
        float rsum[8];
#pragma unroll
        for (int r = 0; r < 8; ++r) {
          float p0 = __expf(s0[r] - mrow[mi][r]);
          float p1 = __expf(s1[r] - mrow[mi][r]);
          rsum[r] = p0 + p1;
          int prow = r + 8 * kh;
          sP[wave * 512 + prow * 32 + m16]      = (_Float16)p0;
          sP[wave * 512 + prow * 32 + 16 + m16] = (_Float16)p1;
        }
#pragma unroll
        for (int r = 0; r < 8; ++r) {
          float t = rsum[r];
          t += __shfl_xor(t, 1);
          t += __shfl_xor(t, 2);
          t += __shfl_xor(t, 4);
          t += __shfl_xor(t, 8);
          lrow[mi][r] = lrow[mi][r] * fs[r] + t;
        }
        __builtin_amdgcn_wave_barrier();  // keep LDS store->load order

        // ---- P A-fragment (C-layout -> A-layout via LDS) ----
        h8 plo = *(const h8*)(sP + wave * 512 + m16 * 32 + kh * 16);
        h8 phi = *(const h8*)(sP + wave * 512 + m16 * 32 + kh * 16 + 8);
        v16h aP = SHUF16(plo, phi);

        // ---- V B-fragments (keys = 32-key pair, N = head-dim halves) ----
        int kb = ntl * 16;
        {
          h8 vlo = *(const h8*)(sVT + m16 * 192 + kb + kh * 16);
          h8 vhi = *(const h8*)(sVT + m16 * 192 + kb + kh * 16 + 8);
          v16h bV = SHUF16(vlo, vhi);
          o0[mi] = wmma_f16f32(aP, bV, o0[mi]);
        }
        {
          h8 vlo = *(const h8*)(sVT + (16 + m16) * 192 + kb + kh * 16);
          h8 vhi = *(const h8*)(sVT + (16 + m16) * 192 + kb + kh * 16 + 8);
          v16h bV = SHUF16(vlo, vhi);
          o1[mi] = wmma_f16f32(aP, bV, o1[mi]);
        }
      }
    }
  }

  // ---- finalize: O /= l, scatter to (b, L, C) f16 ----
#pragma unroll
  for (int mi = 0; mi < 2; ++mi) {
#pragma unroll
    for (int r = 0; r < 8; ++r) {
      float inv = 1.0f / lrow[mi][r];
      int m  = (wave * 2 + mi) * 16 + r + 8 * kh;
      int qy = m >> 4, qx = m & 15;
      size_t rowG = (size_t)b * LTOT + (size_t)(wy * 16 + qy) * 128 + (wx * 16 + qx);
      attn[rowG * CDIM + head * 32 + m16]      = (_Float16)(o0[mi][r] * inv);
      attn[rowG * CDIM + head * 32 + 16 + m16] = (_Float16)(o1[mi][r] * inv);
    }
  }
}

// ---------------------------------------------------------------------------
// FFN elementwise / depthwise conv kernels
// ---------------------------------------------------------------------------
__global__ __launch_bounds__(256) void ffn_gate_k(const _Float16* __restrict__ gv,
                                                  float* __restrict__ h, int total) {
  int i = blockIdx.x * 256 + threadIdx.x;
  if (i >= total) return;
  int pix = i & 16383;
  int ch  = (i >> 14) & 511;
  int b   = i >> 23;
  size_t row = (size_t)b * LTOT + pix;
  float g  = (float)gv[row * 1024 + ch];
  float vv = (float)gv[row * 1024 + 512 + ch];
  float si = g / (1.0f + __expf(-g));
  h[i] = si * vv;
}

__global__ __launch_bounds__(256) void dwconv1_k(const float* __restrict__ h,
                                                 const float* __restrict__ w,
                                                 const float* __restrict__ bias,
                                                 float* __restrict__ out, int total) {
  int i = blockIdx.x * 256 + threadIdx.x;
  if (i >= total) return;
  int x  = i & 127;
  int y  = (i >> 7) & 127;
  int ch = (i >> 14) & 511;
  const float* base = h + (size_t)(i & ~16383);
  float acc = bias[ch];
#pragma unroll
  for (int ty = 0; ty < 3; ++ty) {
    int yy = y + ty - 1;
    if (yy < 0 || yy > 127) continue;
#pragma unroll
    for (int tx = 0; tx < 3; ++tx) {
      int xx = x + tx - 1;
      if (xx < 0 || xx > 127) continue;
      acc += base[yy * 128 + xx] * w[ch * 9 + ty * 3 + tx];
    }
  }
  out[i] = acc;
}

__global__ __launch_bounds__(256) void dwconv2_k(const float* __restrict__ h1,
                                                 const float* __restrict__ w,
                                                 const float* __restrict__ bias,
                                                 const float* __restrict__ scale,
                                                 _Float16* __restrict__ hT, int total) {
  int i = blockIdx.x * 256 + threadIdx.x;
  if (i >= total) return;
  int x  = i & 127;
  int y  = (i >> 7) & 127;
  int ch = (i >> 14) & 511;
  int b  = i >> 23;
  const float* base = h1 + (size_t)(i & ~16383);
  float acc = bias[ch];
#pragma unroll
  for (int ty = 0; ty < 3; ++ty) {
    int yy = y + (ty - 1) * 2;
    if (yy < 0 || yy > 127) continue;
#pragma unroll
    for (int tx = 0; tx < 3; ++tx) {
      int xx = x + (tx - 1) * 2;
      if (xx < 0 || xx > 127) continue;
      acc += base[yy * 128 + xx] * w[ch * 9 + ty * 3 + tx];
    }
  }
  float val = h1[i] + scale[0] * acc;
  size_t row = (size_t)b * LTOT + (y * 128 + x);
  hT[row * 512 + ch] = (_Float16)val;
}

// ---------------------------------------------------------------------------
// Launch
// ---------------------------------------------------------------------------
extern "C" void kernel_launch(void* const* d_in, const int* in_sizes, int n_in,
                              void* d_out, int out_size, void* d_ws, size_t ws_size,
                              hipStream_t stream) {
  (void)in_sizes; (void)n_in; (void)out_size; (void)ws_size;

  const float* x      = (const float*)d_in[0];
  const float* qkv_w  = (const float*)d_in[1];
  const float* qkv_b  = (const float*)d_in[2];
  const float* proj_w = (const float*)d_in[3];
  const float* proj_b = (const float*)d_in[4];
  const float* rpb    = (const float*)d_in[5];
  const float* n1_w   = (const float*)d_in[6];
  const float* n1_b   = (const float*)d_in[7];
  const float* ls1    = (const float*)d_in[8];
  const float* n2_w   = (const float*)d_in[9];
  const float* n2_b   = (const float*)d_in[10];
  const float* ls2    = (const float*)d_in[11];
  const float* fcgv_w = (const float*)d_in[12];
  const float* fcgv_b = (const float*)d_in[13];
  const float* dw_w   = (const float*)d_in[14];
  const float* dw_b   = (const float*)d_in[15];
  const float* dwd_w  = (const float*)d_in[16];
  const float* dwd_b  = (const float*)d_in[17];
  const float* dil_s  = (const float*)d_in[18];
  const float* fco_w  = (const float*)d_in[19];
  const float* fco_b  = (const float*)d_in[20];

  char* ws = (char*)d_ws;
  const size_t OFF_XN  = 4096;                                   // xn / xn2 f16 (12.6 MB)
  const size_t OFF_BIG = OFF_XN + (size_t)32768 * 192 * 2;       // qkv f16 -> gv f16 -> h1 f32 (67 MB)
  const size_t OFF_ATT = OFF_BIG + (size_t)67108864;             // attn f16 (12.6 MB)
  const size_t OFF_X1  = OFF_ATT + (size_t)32768 * 192 * 2;      // x1 f32 (25.2 MB)
  const size_t OFF_H   = OFF_X1 + (size_t)32768 * 192 * 4;       // h f32 -> hT f16 (67 MB)
  const size_t OFF_W   = OFF_H + (size_t)67108864;               // f16 weights (~0.9 MB)

  float*     stats = (float*)ws;
  _Float16*  xn    = (_Float16*)(ws + OFF_XN);
  _Float16*  bigH  = (_Float16*)(ws + OFF_BIG);   // qkv / gv (f16 view)
  float*     bigF  = (float*)(ws + OFF_BIG);      // h1 (f32 view)
  _Float16*  attnB = (_Float16*)(ws + OFF_ATT);
  float*     x1    = (float*)(ws + OFF_X1);
  float*     hbuf  = (float*)(ws + OFF_H);
  _Float16*  hT    = (_Float16*)(ws + OFF_H);

  _Float16* wqkv_h  = (_Float16*)(ws + OFF_W);
  _Float16* wproj_h = wqkv_h  + 576 * 192;   // 110592
  _Float16* wfcgv_h = wproj_h + 192 * 192;   //  36864
  _Float16* wfco_h  = wfcgv_h + 1024 * 192;  // 196608

  const int TOT_XC = 2 * LTOT * CDIM;       // 6,291,456
  const int TOT_H  = 2 * 512 * LTOT;        // 16,777,216

  // ---- one-time weight conversion to f16 ----
  f32_to_f16_k<<<(576 * 192 + 255) / 256, 256, 0, stream>>>(qkv_w, wqkv_h, 576 * 192);
  f32_to_f16_k<<<(192 * 192 + 255) / 256, 256, 0, stream>>>(proj_w, wproj_h, 192 * 192);
  f32_to_f16_k<<<(1024 * 192 + 255) / 256, 256, 0, stream>>>(fcgv_w, wfcgv_h, 1024 * 192);
  f32_to_f16_k<<<(192 * 512 + 255) / 256, 256, 0, stream>>>(fco_w, wfco_h, 192 * 512);

  // ---- LN1 ----
  zero_stats_k<<<1, 32, 0, stream>>>(stats);
  iln_stats_k<<<dim3(256, 2), 256, 0, stream>>>(x, stats, 0);
  iln_apply_k<<<TOT_XC / 256, 256, 0, stream>>>(x, stats, 0, n1_w, n1_b, xn, TOT_XC);

  // ---- QKV projection: (32768,192) @ (192,576) -> f16 ----
  gemm_wmma_k<<<dim3(9, 256), 256, 0, stream>>>(xn, wqkv_h, qkv_b, 576, 192,
                                                bigH, nullptr, nullptr, nullptr,
                                                stats, 0, 0);

  // ---- Overlapping window attention ----
  oca_attn_k<<<768, 256, 0, stream>>>(bigH, rpb, attnB);

  // ---- Output projection + residual: x1 = x + std1*ls1*proj ----
  gemm_wmma_k<<<dim3(3, 256), 256, 0, stream>>>(attnB, wproj_h, proj_b, 192, 192,
                                                nullptr, x1, x, ls1, stats, 0, 1);

  // ---- LN2 ----
  iln_stats_k<<<dim3(256, 2), 256, 0, stream>>>(x1, stats, 4);
  iln_apply_k<<<TOT_XC / 256, 256, 0, stream>>>(x1, stats, 4, n2_w, n2_b, xn, TOT_XC);

  // ---- FFN up-projection (gate+value): (32768,192) @ (192,1024) -> f16 ----
  gemm_wmma_k<<<dim3(16, 256), 256, 0, stream>>>(xn, wfcgv_h, fcgv_b, 1024, 192,
                                                 bigH, nullptr, nullptr, nullptr,
                                                 stats, 0, 0);

  // ---- SiLU gate -> h (b,512,128,128) f32 ----
  ffn_gate_k<<<TOT_H / 256, 256, 0, stream>>>(bigH, hbuf, TOT_H);

  // ---- depthwise 3x3 -> h1 (reuses gv region) ----
  dwconv1_k<<<TOT_H / 256, 256, 0, stream>>>(hbuf, dw_w, dw_b, bigF, TOT_H);

  // ---- dilated depthwise + residual -> hT f16 (b*L, 512) (reuses h region) ----
  dwconv2_k<<<TOT_H / 256, 256, 0, stream>>>(bigF, dwd_w, dwd_b, dil_s, hT, TOT_H);

  // ---- FFN down-projection + residual: out = x1 + std2*ls2*fco ----
  gemm_wmma_k<<<dim3(3, 256), 256, 0, stream>>>(hT, wfco_h, fco_b, 192, 512,
                                                nullptr, (float*)d_out, x1, ls2,
                                                stats, 4, 1);
}
